// negLogLikelihood_per_sample_for_splitting_1589137899800
// MI455X (gfx1250) — compile-verified, matrix operand order checked by reading, not run on hardware
//
#include <hip/hip_runtime.h>
#include <math.h>

typedef __attribute__((ext_vector_type(16))) _Float16 v16h;
typedef __attribute__((ext_vector_type(8)))  _Float16 v8h;
typedef __attribute__((ext_vector_type(4)))  _Float16 v4h;
typedef __attribute__((ext_vector_type(8)))  float    v8f;
typedef __attribute__((ext_vector_type(4)))  float    v4f;

#define STAGE_M 1024   // M-slab staged in LDS per block iteration

// ---------------------------------------------------------------------------
// Kernel 1: h[m] = exp(refer_prediction[m]), split into f16 hi + f16 lo so
// that hi+lo ~ h with ~21 bits of mantissa. Products hi*mask and lo*mask are
// exact (mask is 0/1), so two f32-accumulating WMMAs recover f32 accuracy.
// ---------------------------------------------------------------------------
__global__ void cox_prep_kernel(const float* __restrict__ refer_pred,
                                _Float16* __restrict__ hhi,
                                _Float16* __restrict__ hlo,
                                int M) {
    int m = blockIdx.x * blockDim.x + threadIdx.x;
    if (m < M) {
        float h  = expf(refer_pred[m]);
        _Float16 hi = (_Float16)h;
        _Float16 lo = (_Float16)(h - (float)hi);
        hhi[m] = hi;
        hlo[m] = lo;
    }
}

// ---------------------------------------------------------------------------
// Kernel 2: masked GEMV on the WMMA pipe, LDS-staged.
//   S[n] = sum_m h[m] * (refer_time[m] >= time[n])
//
// Block = 256 threads = 8 wave32s; each wave owns one 16-wide n-tile.
// The block cooperatively stages a 1024-element M-slab (Hhi/Hlo f16 + rt f32,
// 8 KB LDS) once, then every wave runs 32 inner WMMA chunks out of LDS —
// removing the 8x redundant vector-memory streaming of identical data.
//
// A (16x32 f16) = h values replicated across all 16 rows (hi then lo part).
// B (32x16 f16) = 0/1 risk-set mask built from f32 compares.
// D rows are all identical => acc[0] of lanes 0..15 is the answer for n=lane.
//
// A-element layout (ISA 7.12.2, wave32, 16-bit A 16x32):
//   lane L: row M = L%16 ; element e in v16h maps to
//   k = (e/8)*16 + (L/16)*8 + (e%8)
// B mirrors this with lane -> column n. C/D f32: lane L, vgpr v ->
//   (M = v + 8*(L/16), N = L%16).
// ---------------------------------------------------------------------------
__global__ void __launch_bounds__(256)
cox_partial_wmma_kernel(const float* __restrict__ prediction,
                        const float* __restrict__ targets,
                        const float* __restrict__ time_,
                        const float* __restrict__ refer_time,
                        const _Float16* __restrict__ hhi,
                        const _Float16* __restrict__ hlo,
                        float* __restrict__ out,
                        int N, int M) {
    __shared__ __align__(16) _Float16 sHhi[STAGE_M];  // 2 KB
    __shared__ __align__(16) _Float16 sHlo[STAGE_M];  // 2 KB
    __shared__ __align__(16) float    sRt [STAGE_M];  // 4 KB

    const int tid    = threadIdx.x;
    const int wave   = tid >> 5;                    // 0..7
    const int lane   = tid & 31;                    // wave32 lane
    const int tile   = blockIdx.x * 8 + wave;       // 16-wide n tile
    const int n_base = tile * 16;
    const int n      = lane & 15;                   // column this lane owns
    const int hgrp   = lane >> 4;                   // half-wave group (0/1)

    const float t_n = time_[n_base + n];            // threshold for column n

    v8f acc = {};

    for (int stage = 0; stage < M; stage += STAGE_M) {
        // ---- cooperative stage: 4 elements of each array per thread ----
        const int s4 = tid * 4;                              // 0..1020
        v4f rt_reg = *(const v4f*)(refer_time + stage + s4); // 16 B
        v4h hh_reg = *(const v4h*)(hhi        + stage + s4); // 8 B
        v4h hl_reg = *(const v4h*)(hlo        + stage + s4); // 8 B

        __syncthreads();   // previous slab fully consumed before overwrite
        *(v4f*)(sRt  + s4) = rt_reg;
        *(v4h*)(sHhi + s4) = hh_reg;
        *(v4h*)(sHlo + s4) = hl_reg;
        __syncthreads();   // slab visible to all waves

        // ---- 32 inner WMMA chunks (32 m each) out of LDS ----
        for (int c = 0; c < STAGE_M; c += 32) {
            const int base = c + hgrp * 8;          // k-range this half-wave supplies

            // A operands: h split into hi/lo f16, two 8-wide k runs 16 apart
            v8h ah0 = *(const v8h*)(sHhi + base);        // k = base .. base+7
            v8h ah1 = *(const v8h*)(sHhi + base + 16);   // k = base+16 .. base+23
            v8h al0 = *(const v8h*)(sHlo + base);
            v8h al1 = *(const v8h*)(sHlo + base + 16);
            v16h Ahi = __builtin_shufflevector(ah0, ah1,
                        0,1,2,3,4,5,6,7,8,9,10,11,12,13,14,15);
            v16h Alo = __builtin_shufflevector(al0, al1,
                        0,1,2,3,4,5,6,7,8,9,10,11,12,13,14,15);

            // B operand: risk-set mask (refer_time[m] >= t_n)
            v4f r0 = *(const v4f*)(sRt + base);
            v4f r1 = *(const v4f*)(sRt + base + 4);
            v4f r2 = *(const v4f*)(sRt + base + 16);
            v4f r3 = *(const v4f*)(sRt + base + 20);

            v16h Bm;
#pragma unroll
            for (int j = 0; j < 4; ++j) {
                Bm[j]      = (r0[j] >= t_n) ? (_Float16)1.0f : (_Float16)0.0f;
                Bm[4 + j]  = (r1[j] >= t_n) ? (_Float16)1.0f : (_Float16)0.0f;
                Bm[8 + j]  = (r2[j] >= t_n) ? (_Float16)1.0f : (_Float16)0.0f;
                Bm[12 + j] = (r3[j] >= t_n) ? (_Float16)1.0f : (_Float16)0.0f;
            }

            // Two WMMAs share one mask tile: hi part, then lo correction.
            acc = __builtin_amdgcn_wmma_f32_16x16x32_f16(
                      false, Ahi, false, Bm, (short)0, acc, false, false);
            acc = __builtin_amdgcn_wmma_f32_16x16x32_f16(
                      false, Alo, false, Bm, (short)0, acc, false, false);
        }
    }

    // Every D row is identical; lanes 0..15 hold row 0 of columns 0..15 in acc[0].
    if (hgrp == 0) {
        const int idx = n_base + n;
        if (idx < N) {
            float risk    = prediction[idx];
            float hr      = expf(risk);
            float partial = hr + acc[0];
            float uncens  = risk - logf(partial);
            out[idx]      = -uncens * targets[idx];
        }
    }
}

// ---------------------------------------------------------------------------
// Host-side launch. Inputs (setup_inputs order):
//   d_in[0] prediction [N], d_in[1] targets [N], d_in[2] time [N],
//   d_in[3] refer_prediction [M], d_in[4] refer_time [M]. Output: [N] f32.
// d_ws: Hhi (M f16) followed by Hlo (M f16) = 64 KB.
// ---------------------------------------------------------------------------
extern "C" void kernel_launch(void* const* d_in, const int* in_sizes, int n_in,
                              void* d_out, int out_size, void* d_ws, size_t ws_size,
                              hipStream_t stream) {
    const float* prediction = (const float*)d_in[0];
    const float* targets    = (const float*)d_in[1];
    const float* time_      = (const float*)d_in[2];
    const float* refer_pred = (const float*)d_in[3];
    const float* refer_time = (const float*)d_in[4];
    float* out = (float*)d_out;

    const int N = in_sizes[0];
    const int M = in_sizes[3];

    _Float16* hhi = (_Float16*)d_ws;
    _Float16* hlo = hhi + M;

    // 1) exp + hi/lo split of the reference hazards
    {
        dim3 block(256);
        dim3 grid((M + 255) / 256);
        cox_prep_kernel<<<grid, block, 0, stream>>>(refer_pred, hhi, hlo, M);
    }

    // 2) WMMA masked GEMV (LDS-staged) + per-sample likelihood epilogue
    {
        dim3 block(256);                    // 8 wave32s -> 128 n per block
        dim3 grid((N + 127) / 128);
        cox_partial_wmma_kernel<<<grid, block, 0, stream>>>(
            prediction, targets, time_, refer_time, hhi, hlo, out, N, M);
    }
}
